// RefillModelRNNSwitch_29343216566609
// MI455X (gfx1250) — matrix operand order repaired and checked by reading, not compiled, not run on hardware
//
#include <hip/hip_runtime.h>
#include <math.h>

// ---- problem constants (match reference) ----
#define V_   32000
#define E_   512
#define M_   16
#define B_   16
#define T_   256
#define TY_  64

#define SV        16                 // vocab splits for the CE GEMM
#define CT_TOTAL  (V_/16)            // 2000 col tiles of 16
#define CT_PER_SV (CT_TOTAL/SV)      // 125
#define APAD      516                // padded f32 LDS row pitch
#define BPITCH    520                // padded bf16 LDS row pitch (conflict-free frags)

typedef __attribute__((ext_vector_type(2)))  float  v2f;
typedef __attribute__((ext_vector_type(8)))  float  v8f;
typedef __attribute__((ext_vector_type(8)))  __bf16 v8bf;
typedef __attribute__((ext_vector_type(16))) __bf16 v16bf;

// D = A(16x4 f32) * B(4x16 f32) + C(16x16 f32)
__device__ __forceinline__ v8f wmma_f32(v2f a, v2f b, v8f c) {
  return __builtin_amdgcn_wmma_f32_16x16x4_f32(false, a, false, b, (short)0, c,
                                               false, false);
}
// D = A(16x32 bf16) * B(32x16 bf16) + C(16x16 f32)
__device__ __forceinline__ v8f wmma_bf16(v16bf a, v16bf b, v8f c) {
  return __builtin_amdgcn_wmma_f32_16x16x32_bf16(false, a, false, b, (short)0, c,
                                                 false, false);
}

__device__ __forceinline__ float wave_reduce_sum(float v) {
  #pragma unroll
  for (int off = 16; off; off >>= 1) v += __shfl_xor(v, off, 32);
  return v;
}

// -------------------------------------------------------------------------
// Kernel 0: split-precision decomposition  x ~= hi(bf16) + lo(bf16)
// -------------------------------------------------------------------------
__global__ void k_split_bf16(const float* __restrict__ src, __bf16* __restrict__ hi,
                             __bf16* __restrict__ lo, int n) {
  int i = blockIdx.x * 256 + threadIdx.x;
  if (i < n) {
    float x = src[i];
    __bf16 h = (__bf16)x;
    hi[i] = h;
    lo[i] = (__bf16)(x - (float)h);
  }
}

// -------------------------------------------------------------------------
// Kernel 1: ye = norm(embed[y]);  ze = norm(embed[z])   (row-wise /std, ddof=1)
// -------------------------------------------------------------------------
__global__ void k_embed_norm(const int* __restrict__ yidx, const int* __restrict__ zidx,
                             const float* __restrict__ embed,
                             float* __restrict__ ye, float* __restrict__ ze) {
  int row = blockIdx.x;
  const float* src;
  float* dst;
  if (row < B_ * TY_) {
    src = embed + (size_t)yidx[row] * E_;
    dst = ye + (size_t)row * E_;
  } else {
    int r = row - B_ * TY_;
    src = embed + (size_t)zidx[r] * E_;
    dst = ze + (size_t)r * E_;
  }
  __shared__ float red1[256], red2[256];
  int tid = threadIdx.x;
  float a = src[tid], b = src[tid + 256];
  red1[tid] = a + b;
  red2[tid] = a * a + b * b;
  __syncthreads();
  for (int off = 128; off; off >>= 1) {
    if (tid < off) { red1[tid] += red1[tid + off]; red2[tid] += red2[tid + off]; }
    __syncthreads();
  }
  float S1 = red1[0], S2 = red2[0];
  float var = (S2 - S1 * S1 / (float)E_) / (float)(E_ - 1);
  float sc = 1.0f / (1e-5f + sqrtf(fmaxf(var, 0.f)));
  dst[tid] = a * sc;
  dst[tid + 256] = b * sc;
}

// -------------------------------------------------------------------------
// Kernel 2: yk = ye @ selk_w^T + selk_b     (1024 x 512) @ (512 x 512), f32 WMMA
// -------------------------------------------------------------------------
__global__ void k_yk(const float* __restrict__ ye, const float* __restrict__ selk_w,
                     const float* __restrict__ selk_b, float* __restrict__ yk) {
  int rt = blockIdx.x, ct = blockIdx.y;
  int lane = threadIdx.x & 31;
  int half = lane >> 4, mn = lane & 15, koff = 2 * half;
  const float* pa = ye + (size_t)(rt * 16 + mn) * E_;
  const float* pb = selk_w + (size_t)(ct * 16 + mn) * E_;  // B[k,e]=selk_w[e*E+k]
  v8f acc = {};
  for (int kt = 0; kt < E_ / 4; ++kt) {
    v2f a = *(const v2f*)(pa + kt * 4 + koff);
    v2f b = *(const v2f*)(pb + kt * 4 + koff);
    acc = wmma_f32(a, b, acc);
  }
  float bias = selk_b[ct * 16 + mn];
  #pragma unroll
  for (int j = 0; j < 8; ++j) {
    int r = rt * 16 + j + 8 * half;
    yk[(size_t)r * E_ + ct * 16 + mn] = acc[j] + bias;
  }
}

// -------------------------------------------------------------------------
// Kernel 3: attn = softmax_t(selq_w . yk); cand[b,m,:] = sum_t attn * ye
// -------------------------------------------------------------------------
__global__ void k_attn_cand(const float* __restrict__ yk, const float* __restrict__ selq_w,
                            const float* __restrict__ ye, float* __restrict__ cand) {
  int b = blockIdx.x, m = blockIdx.y;
  __shared__ float s_attn[TY_];
  int tid = threadIdx.x, lane = tid & 31, w = tid >> 5;
  const float* q = selq_w + (size_t)m * E_;
  for (int ti = 0; ti < 8; ++ti) {
    int t = w * 8 + ti;
    const float* k = yk + (size_t)(b * TY_ + t) * E_;
    float p = 0.f;
    for (int i = lane; i < E_; i += 32) p += q[i] * k[i];
    p = wave_reduce_sum(p);
    if (lane == 0) s_attn[t] = p;
  }
  __syncthreads();
  if (tid == 0) {
    float mx = -INFINITY;
    for (int t = 0; t < TY_; ++t) mx = fmaxf(mx, s_attn[t]);
    float s = 0.f;
    for (int t = 0; t < TY_; ++t) { float e = __expf(s_attn[t] - mx); s_attn[t] = e; s += e; }
    float inv = 1.f / s;
    for (int t = 0; t < TY_; ++t) s_attn[t] *= inv;
  }
  __syncthreads();
  for (int e = tid; e < E_; e += 256) {
    float v = 0.f;
    for (int t = 0; t < TY_; ++t) v += s_attn[t] * ye[(size_t)(b * TY_ + t) * E_ + e];
    cand[(size_t)(b * M_ + m) * E_ + e] = v;
  }
}

// -------------------------------------------------------------------------
// Kernel 4: sequential scan over T steps (single 32-wave workgroup, f32 WMMA)
// -------------------------------------------------------------------------
__global__ void __launch_bounds__(1024) k_scan(
    const float* __restrict__ ze, const float* __restrict__ cand,
    const float* __restrict__ init_w, const float* __restrict__ trans_w,
    const float* __restrict__ trans_b, const float* __restrict__ upd_w,
    const float* __restrict__ upd_b, const float* __restrict__ sel_w,
    const float* __restrict__ sel_b, float* __restrict__ fs) {
  __shared__ float s_xs[B_ * APAD];
  __shared__ float s_pre[B_ * APAD];
  __shared__ float s_sel[B_ * M_];
  __shared__ float s_red[2];
  int tid = threadIdx.x, lane = tid & 31, w = tid >> 5;
  int half = lane >> 4, mn = lane & 15, koff = 2 * half;

  // xs(0) = broadcast(norm(init_w[:,0]))
  if (tid == 0) { s_red[0] = 0.f; s_red[1] = 0.f; }
  __syncthreads();
  float v0 = 0.f;
  if (tid < E_) {
    v0 = init_w[(size_t)tid * M_];
    atomicAdd(&s_red[0], v0);
    atomicAdd(&s_red[1], v0 * v0);
  }
  __syncthreads();
  if (tid < E_) {
    float S1 = s_red[0], S2 = s_red[1];
    float var = (S2 - S1 * S1 / (float)E_) / (float)(E_ - 1);
    float sc = 1.f / (1e-5f + sqrtf(fmaxf(var, 0.f)));
    float nv = v0 * sc;
    for (int b = 0; b < B_; ++b) s_xs[b * APAD + tid] = nv;
  }
  __syncthreads();

  for (int t = 0; t < T_; ++t) {
    // ---- Phase A: pre = xs + xs@trans_w^T + xz@upd_w^T + biases (WMMA) ----
    {
      int et = w;  // 32 waves <-> 32 column tiles of E
      const float* pb1 = trans_w + (size_t)(et * 16 + mn) * E_;
      const float* pb2 = upd_w + (size_t)(et * 16 + mn) * E_;
      const float* pa2 = ze + (size_t)(mn * T_ + t) * E_;  // xz rows
      const float* paL = &s_xs[mn * APAD];
      v8f acc = {};
      for (int kt = 0; kt < E_ / 4; ++kt) {
        int k0 = kt * 4 + koff;
        v2f a1 = *(const v2f*)(paL + k0);
        v2f b1 = *(const v2f*)(pb1 + k0);
        acc = wmma_f32(a1, b1, acc);
        v2f a2 = *(const v2f*)(pa2 + k0);
        v2f b2 = *(const v2f*)(pb2 + k0);
        acc = wmma_f32(a2, b2, acc);
      }
      int e = et * 16 + mn;
      float bias = trans_b[e] + upd_b[e];
      #pragma unroll
      for (int j = 0; j < 8; ++j) {
        int r = j + 8 * half;
        s_pre[r * APAD + e] = acc[j] + s_xs[r * APAD + e] + bias;
      }
    }
    __syncthreads();
    // ---- Phase B: row norm -> s_xs ----
    if (w < B_) {
      int b = w;
      float s1 = 0.f, s2 = 0.f;
      for (int i = lane; i < E_; i += 32) { float vv = s_pre[b * APAD + i]; s1 += vv; s2 += vv * vv; }
      s1 = wave_reduce_sum(s1);
      s2 = wave_reduce_sum(s2);
      float var = (s2 - s1 * s1 / (float)E_) / (float)(E_ - 1);
      float sc = 1.f / (1e-5f + sqrtf(fmaxf(var, 0.f)));
      for (int i = lane; i < E_; i += 32) s_xs[b * APAD + i] = s_pre[b * APAD + i] * sc;
    }
    __syncthreads();
    // ---- Phase C: sel = softmax_m(xs @ sel_w^T + sel_b) ----
    if (tid < B_ * M_) {
      int b = tid >> 4, m = tid & 15;
      const float* wrow = sel_w + (size_t)m * E_;
      float p = sel_b[m];
      for (int i = 0; i < E_; ++i) p += s_xs[b * APAD + i] * wrow[i];
      s_sel[tid] = p;
    }
    __syncthreads();
    if (tid < B_) {
      float mx = -INFINITY;
      for (int m = 0; m < M_; ++m) mx = fmaxf(mx, s_sel[tid * M_ + m]);
      float s = 0.f;
      for (int m = 0; m < M_; ++m) { float e = __expf(s_sel[tid * M_ + m] - mx); s_sel[tid * M_ + m] = e; s += e; }
      float inv = 1.f / s;
      for (int m = 0; m < M_; ++m) s_sel[tid * M_ + m] *= inv;
    }
    __syncthreads();
    // ---- Phase D: fs[t] = sel[0]*xz + sum_{m>=1} sel[m]*cand[m] ----
    for (int idx = tid; idx < B_ * E_; idx += 1024) {
      int b = idx >> 9, e = idx & 511;
      float val = s_sel[b * M_] * ze[(size_t)(b * T_ + t) * E_ + e];
      #pragma unroll
      for (int m = 1; m < M_; ++m) val += s_sel[b * M_ + m] * cand[(size_t)(b * M_ + m) * E_ + e];
      fs[(size_t)(t * B_ + b) * E_ + e] = val;
    }
    __syncthreads();
  }
}

// -------------------------------------------------------------------------
// Kernel 5: fsn[b,t,:] = norm(fs[t,b,:])  (transpose + row norm)
// -------------------------------------------------------------------------
__global__ void k_norm_fs(const float* __restrict__ fs, float* __restrict__ fsn) {
  int r = blockIdx.x;  // b*T + t
  int b = r / T_, t = r % T_;
  const float* src = fs + (size_t)(t * B_ + b) * E_;
  float* dst = fsn + (size_t)r * E_;
  __shared__ float red1[256], red2[256];
  int tid = threadIdx.x;
  float a = src[tid], bb = src[tid + 256];
  red1[tid] = a + bb;
  red2[tid] = a * a + bb * bb;
  __syncthreads();
  for (int off = 128; off; off >>= 1) {
    if (tid < off) { red1[tid] += red1[tid + off]; red2[tid] += red2[tid + off]; }
    __syncthreads();
  }
  float S1 = red1[0], S2 = red2[0];
  float var = (S2 - S1 * S1 / (float)E_) / (float)(E_ - 1);
  float sc = 1.0f / (1e-5f + sqrtf(fmaxf(var, 0.f)));
  dst[tid] = a * sc;
  dst[tid + 256] = bb * sc;
}

// -------------------------------------------------------------------------
// Kernel 6: fused logits GEMM + online logsumexp over the vocab.
// bf16x2 split-precision: logits = Ahi*Bhi + Ahi*Blo + Alo*Bhi (fp32 accum).
// grid = (128 row-groups of 32 rows, SV vocab splits), 256 threads (8 waves).
// -------------------------------------------------------------------------
__global__ void __launch_bounds__(256) k_ce_lse(
    const __bf16* __restrict__ fsn_hi, const __bf16* __restrict__ fsn_lo,
    const __bf16* __restrict__ emb_hi, const __bf16* __restrict__ emb_lo,
    float* __restrict__ partials) {
  int rg = blockIdx.x;  // 32-row group
  int sv = blockIdx.y;  // vocab split
  __shared__ __bf16 s_Ah[32 * BPITCH];
  __shared__ __bf16 s_Al[32 * BPITCH];
  __shared__ float s_part[8][32][2];
  int tid = threadIdx.x, lane = tid & 31, w = tid >> 5;
  int half = lane >> 4, mn = lane & 15;

  for (int i = tid; i < 32 * E_; i += 256) {
    int r = i >> 9, e = i & 511;
    s_Ah[r * BPITCH + e] = fsn_hi[(size_t)(rg * 32 + r) * E_ + e];
    s_Al[r * BPITCH + e] = fsn_lo[(size_t)(rg * 32 + r) * E_ + e];
  }
  __syncthreads();

  float mArr[16], sArr[16];
  #pragma unroll
  for (int j = 0; j < 16; ++j) { mArr[j] = -INFINITY; sArr[j] = 0.f; }

  // per-lane A fragment bases (bf16 A 16x32: K in {16h..16h+15})
  const __bf16* pah0 = &s_Ah[mn * BPITCH + 16 * half];
  const __bf16* pal0 = &s_Al[mn * BPITCH + 16 * half];
  const __bf16* pah1 = pah0 + 16 * BPITCH;
  const __bf16* pal1 = pal0 + 16 * BPITCH;

  for (int ct = sv * CT_PER_SV + w; ct < (sv + 1) * CT_PER_SV; ct += 8) {
    // bf16 B 32x16: per lane 16 contiguous K at offset 16*half, col = ct*16+mn
    const __bf16* pbh = emb_hi + (size_t)(ct * 16 + mn) * E_ + 16 * half;
    const __bf16* pbl = emb_lo + (size_t)(ct * 16 + mn) * E_ + 16 * half;
    // prefetch next column tile for this wave (ct+8)
    __builtin_prefetch(pbh + (size_t)8 * 16 * E_, 0, 1);
    __builtin_prefetch(pbl + (size_t)8 * 16 * E_, 0, 1);
    v8f acc0 = {}, acc1 = {};
    for (int kb = 0; kb < E_; kb += 32) {
      v16bf bh = *(const v16bf*)(pbh + kb);
      v16bf bl = *(const v16bf*)(pbl + kb);
      v8bf h0a = *(const v8bf*)(pah0 + kb);
      v8bf h0b = *(const v8bf*)(pah0 + kb + 8);
      v16bf ah0 = __builtin_shufflevector(h0a, h0b, 0,1,2,3,4,5,6,7,8,9,10,11,12,13,14,15);
      v8bf l0a = *(const v8bf*)(pal0 + kb);
      v8bf l0b = *(const v8bf*)(pal0 + kb + 8);
      v16bf al0 = __builtin_shufflevector(l0a, l0b, 0,1,2,3,4,5,6,7,8,9,10,11,12,13,14,15);
      v8bf h1a = *(const v8bf*)(pah1 + kb);
      v8bf h1b = *(const v8bf*)(pah1 + kb + 8);
      v16bf ah1 = __builtin_shufflevector(h1a, h1b, 0,1,2,3,4,5,6,7,8,9,10,11,12,13,14,15);
      v8bf l1a = *(const v8bf*)(pal1 + kb);
      v8bf l1b = *(const v8bf*)(pal1 + kb + 8);
      v16bf al1 = __builtin_shufflevector(l1a, l1b, 0,1,2,3,4,5,6,7,8,9,10,11,12,13,14,15);
      acc0 = wmma_bf16(ah0, bh, acc0);
      acc0 = wmma_bf16(ah0, bl, acc0);
      acc0 = wmma_bf16(al0, bh, acc0);
      acc1 = wmma_bf16(ah1, bh, acc1);
      acc1 = wmma_bf16(ah1, bl, acc1);
      acc1 = wmma_bf16(al1, bh, acc1);
    }
    // online logsumexp update (rows M = j + 8*half per tile)
    #pragma unroll
    for (int j = 0; j < 8; ++j) {
      float l0 = acc0[j];
      float m0 = fmaxf(mArr[j], l0);
      sArr[j] = sArr[j] * __expf(mArr[j] - m0) + __expf(l0 - m0);
      mArr[j] = m0;
      float l1 = acc1[j];
      float m1 = fmaxf(mArr[j + 8], l1);
      sArr[j + 8] = sArr[j + 8] * __expf(mArr[j + 8] - m1) + __expf(l1 - m1);
      mArr[j + 8] = m1;
    }
  }
  // combine across the 16 lanes holding the same row (different vocab cols)
  #pragma unroll
  for (int j = 0; j < 16; ++j) {
    float m = mArr[j], s = sArr[j];
    #pragma unroll
    for (int off = 8; off; off >>= 1) {
      float mo = __shfl_xor(m, off, 32);
      float so = __shfl_xor(s, off, 32);
      float mc = fmaxf(m, mo);
      s = s * __expf(m - mc) + so * __expf(mo - mc);
      m = mc;
    }
    mArr[j] = m; sArr[j] = s;
  }
  if (mn == 0) {
    #pragma unroll
    for (int j = 0; j < 8; ++j) {
      int lr0 = j + 8 * half;
      int lr1 = 16 + j + 8 * half;
      s_part[w][lr0][0] = mArr[j];     s_part[w][lr0][1] = sArr[j];
      s_part[w][lr1][0] = mArr[j + 8]; s_part[w][lr1][1] = sArr[j + 8];
    }
  }
  __syncthreads();
  if (tid < 32) {
    float m = s_part[0][tid][0], s = s_part[0][tid][1];
    for (int ww = 1; ww < 8; ++ww) {
      float mo = s_part[ww][tid][0], so = s_part[ww][tid][1];
      float mc = fmaxf(m, mo);
      s = s * __expf(m - mc) + so * __expf(mo - mc);
      m = mc;
    }
    size_t row = (size_t)rg * 32 + tid;
    partials[row * (2 * SV) + 2 * sv] = m;
    partials[row * (2 * SV) + 2 * sv + 1] = s;
  }
}

// -------------------------------------------------------------------------
// Kernel 7: picked[row] = fsn[row] . embed[x[row]]  (fp32, one wave per row)
// -------------------------------------------------------------------------
__global__ void k_picked(const float* __restrict__ fsn, const float* __restrict__ embed,
                         const int* __restrict__ x, float* __restrict__ picked) {
  int g = blockIdx.x * 8 + (threadIdx.x >> 5);
  int lane = threadIdx.x & 31;
  const float* a = fsn + (size_t)g * E_;
  const float* bvec = embed + (size_t)x[g] * E_;
  float p = 0.f;
  for (int i = lane; i < E_; i += 32) p += a[i] * bvec[i];
  p = wave_reduce_sum(p);
  if (lane == 0) picked[g] = p;
}

// -------------------------------------------------------------------------
// Kernel 8: combine SV partials -> lse; out[b] = -mean_t(picked - lse)
// -------------------------------------------------------------------------
__global__ void k_loss(const float* __restrict__ partials, const float* __restrict__ picked,
                       float* __restrict__ out) {
  int b = blockIdx.x, tid = threadIdx.x;
  size_t row = (size_t)b * T_ + tid;
  float m = partials[row * (2 * SV)], s = partials[row * (2 * SV) + 1];
  for (int i = 1; i < SV; ++i) {
    float mo = partials[row * (2 * SV) + 2 * i], so = partials[row * (2 * SV) + 2 * i + 1];
    float mc = fmaxf(m, mo);
    s = s * __expf(m - mc) + so * __expf(mo - mc);
    m = mc;
  }
  float lse = m + __logf(s);
  float cent = picked[row] - lse;
  __shared__ float red[256];
  red[tid] = cent;
  __syncthreads();
  for (int off = 128; off; off >>= 1) {
    if (tid < off) red[tid] += red[tid + off];
    __syncthreads();
  }
  if (tid == 0) out[b] = -red[0] / (float)T_;
}

// -------------------------------------------------------------------------
extern "C" void kernel_launch(void* const* d_in, const int* in_sizes, int n_in,
                              void* d_out, int out_size, void* d_ws, size_t ws_size,
                              hipStream_t stream) {
  // inputs in setup_inputs() order
  const int* x = (const int*)d_in[1];
  const int* y = (const int*)d_in[2];
  const int* z = (const int*)d_in[3];
  const float* embed_w = (const float*)d_in[4];
  const float* sel_w   = (const float*)d_in[5];
  const float* sel_b   = (const float*)d_in[6];
  const float* selq_w  = (const float*)d_in[7];
  const float* selk_w  = (const float*)d_in[8];
  const float* selk_b  = (const float*)d_in[9];
  const float* init_w  = (const float*)d_in[10];
  const float* trans_w = (const float*)d_in[11];
  const float* trans_b = (const float*)d_in[12];
  const float* upd_w   = (const float*)d_in[13];
  const float* upd_b   = (const float*)d_in[14];
  float* out = (float*)d_out;

  // workspace layout
  float* ws = (float*)d_ws;
  float* ye       = ws;                          // B*TY*E      = 524288 f
  float* ze       = ye + (size_t)B_ * TY_ * E_;  // B*T*E       = 2097152 f
  float* yk       = ze + (size_t)B_ * T_ * E_;   // B*TY*E      = 524288 f
  float* cand     = yk + (size_t)B_ * TY_ * E_;  // B*M*E       = 131072 f
  float* fs       = cand + (size_t)B_ * M_ * E_; // T*B*E       = 2097152 f
  float* fsn      = fs + (size_t)T_ * B_ * E_;   // B*T*E       = 2097152 f
  float* partials = fsn + (size_t)B_ * T_ * E_;  // 4096*2*SV   = 131072 f
  float* picked   = partials + (size_t)B_ * T_ * 2 * SV;  // 4096 f
  // bf16 region (starts 32B-aligned: 30,425,088 bytes of floats above)
  __bf16* fsn_hi = (__bf16*)(picked + B_ * T_);
  __bf16* fsn_lo = fsn_hi + (size_t)B_ * T_ * E_;
  __bf16* emb_hi = fsn_lo + (size_t)B_ * T_ * E_;
  __bf16* emb_lo = emb_hi + (size_t)V_ * E_;

  const int nEmb = V_ * E_;      // 16,384,000
  const int nFsn = B_ * T_ * E_; // 2,097,152

  k_split_bf16<<<(nEmb + 255) / 256, 256, 0, stream>>>(embed_w, emb_hi, emb_lo, nEmb);
  k_embed_norm<<<B_ * TY_ + B_ * T_, 256, 0, stream>>>(y, z, embed_w, ye, ze);
  k_yk<<<dim3((B_ * TY_) / 16, E_ / 16), 32, 0, stream>>>(ye, selk_w, selk_b, yk);
  k_attn_cand<<<dim3(B_, M_), 256, 0, stream>>>(yk, selq_w, ye, cand);
  k_scan<<<1, 1024, 0, stream>>>(ze, cand, init_w, trans_w, trans_b,
                                 upd_w, upd_b, sel_w, sel_b, fs);
  k_norm_fs<<<B_ * T_, 256, 0, stream>>>(fs, fsn);
  k_split_bf16<<<(nFsn + 255) / 256, 256, 0, stream>>>(fsn, fsn_hi, fsn_lo, nFsn);
  k_ce_lse<<<dim3((B_ * T_) / 32, SV), 256, 0, stream>>>(fsn_hi, fsn_lo, emb_hi, emb_lo, partials);
  k_picked<<<(B_ * T_) / 8, 256, 0, stream>>>(fsn, embed_w, x, picked);
  k_loss<<<B_, 256, 0, stream>>>(partials, picked, out);
}
